// GLALayerFused_62302795596484
// MI455X (gfx1250) — matrix operand order repaired and checked by reading, not compiled, not run on hardware
//
#include <hip/hip_runtime.h>
#include <stdint.h>

// ---------------------------------------------------------------------------
// GLA layer, MI455X (gfx1250): bf16 WMMA pipeline.
//   B=4, T=8192, D=1024, CHUNK=32, M = B*T = 32768.
// ws layout (bytes), total ~394MB:
//   xb   : bf16 x           32768*1024*2 = 67,108,864
//   Wb[5]: bf16 weights     5 * 2,097,152
//   Qb,Kb,Vb,Gb             4 * 67,108,864
//   Zb   : gate*O bf16      67,108,864
// ---------------------------------------------------------------------------

#define USE_ASYNC 1

typedef __attribute__((ext_vector_type(16))) __bf16 v16bf;
typedef __attribute__((ext_vector_type(8)))  __bf16 v8bf;
typedef __attribute__((ext_vector_type(4)))  __bf16 v4bf;
typedef __attribute__((ext_vector_type(8)))  float  v8f;

union Frag { v16bf v; v8bf h[2]; };

static constexpr int D  = 1024;   // model dim (= N = K of every GEMM)
static constexpr int M  = 32768;  // B*T rows
static constexpr int LD = 40;     // LDS row stride in bf16 (32 + 8 pad, 16B aligned)

// 16-byte global -> LDS copy. Async (CDNA5 ASYNCcnt) or direct fallback.
__device__ __forceinline__ void cp16(__bf16* dst_lds, const __bf16* src_g) {
#if USE_ASYNC
  unsigned           l = (unsigned)(uintptr_t)dst_lds;            // addr[31:0] == LDS offset
  unsigned long long g = (unsigned long long)(uintptr_t)src_g;
  asm volatile("global_load_async_to_lds_b128 %0, %1, off"
               :: "v"(l), "v"(g) : "memory");
#else
  *(uint4*)dst_lds = *(const uint4*)src_g;
#endif
}

#if USE_ASYNC
#define WAIT_ASYNC(n) asm volatile("s_wait_asynccnt %0" :: "i"(n))
#else
#define WAIT_ASYNC(n)
#endif

__device__ __forceinline__ v8f wmma_bf16(const Frag& a, const Frag& b, v8f c) {
  return __builtin_amdgcn_wmma_f32_16x16x32_bf16(false, a.v, false, b.v,
                                                 (short)0, c, false, false);
}

// ---------------------------------------------------------------------------
// f32 -> bf16 bulk convert (n multiple of 4)
// ---------------------------------------------------------------------------
__global__ __launch_bounds__(256) void cvt_bf16(const float* __restrict__ in,
                                                __bf16* __restrict__ out, int n) {
  int i = (blockIdx.x * 256 + threadIdx.x) * 4;
  if (i < n) {
    float4 v = *(const float4*)(in + i);
    v4bf o = { (__bf16)v.x, (__bf16)v.y, (__bf16)v.z, (__bf16)v.w };
    *(v4bf*)(out + i) = o;
  }
}

// ---------------------------------------------------------------------------
// Tiled GEMM:  Out[m,n] = epi( sum_k A[m,k] * W[n,k] )     (A:[M,D], W:[D,D])
// Block tile 128x128, K-step 32, 8 waves of 32x64, double-buffered async LDS.
// epi: 0 -> bf16 store, 1 -> sigmoid+bf16 store, 2 -> +resid, f32 store
// ---------------------------------------------------------------------------
__global__ __launch_bounds__(256) void gemm128(const __bf16* __restrict__ A,
                                               const __bf16* __restrict__ W,
                                               const float* __restrict__ resid,
                                               void* __restrict__ Out, int epi) {
  __shared__ __bf16 Abuf[2][128 * LD];
  __shared__ __bf16 Bbuf[2][128 * LD];

  const int t    = threadIdx.x;
  const int lane = t & 31;
  const int w    = t >> 5;
  const int bM   = blockIdx.y * 128;
  const int bN   = blockIdx.x * 128;
  const int mBase = (w & 3) * 32;   // 2 M-tiles of 16
  const int nBase = (w >> 2) * 64;  // 4 N-tiles of 16

  const int rowS = t >> 2;          // staging row 0..63 (and +64)
  const int chk  = t & 3;           // 16B chunk within a 64B (32-bf16) row

  v8f acc[2][4] = {};

  auto issue = [&](int kt, int b) {
    const __bf16* ga = A + (size_t)(bM + rowS) * D + kt * 32 + chk * 8;
    cp16(&Abuf[b][rowS * LD + chk * 8], ga);
    cp16(&Abuf[b][(rowS + 64) * LD + chk * 8], ga + (size_t)64 * D);
    const __bf16* gb = W + (size_t)(bN + rowS) * D + kt * 32 + chk * 8;
    cp16(&Bbuf[b][rowS * LD + chk * 8], gb);
    cp16(&Bbuf[b][(rowS + 64) * LD + chk * 8], gb + (size_t)64 * D);
  };

  issue(0, 0);

  const int hk = (lane >> 4) * 8;   // A frag K origin per half-wave
  const int ks = (lane >> 4) * 16;  // unused; B frag also uses hk
  const int l16 = lane & 15;

  for (int kt = 0; kt < D / 32; ++kt) {
    const int cur = kt & 1;
    if (kt + 1 < D / 32) { issue(kt + 1, cur ^ 1); WAIT_ASYNC(4); }
    else                 { WAIT_ASYNC(0); }
    __syncthreads();

    Frag a[2], b[4];
    #pragma unroll
    for (int tm = 0; tm < 2; ++tm) {
      const __bf16* p = &Abuf[cur][(mBase + tm * 16 + l16) * LD];
      a[tm].h[0] = *(const v8bf*)(p + hk);
      a[tm].h[1] = *(const v8bf*)(p + hk + 16);
    }
    #pragma unroll
    for (int tn = 0; tn < 4; ++tn) {
      const __bf16* p = &Bbuf[cur][(nBase + tn * 16 + l16) * LD];
      b[tn].h[0] = *(const v8bf*)(p + hk);
      b[tn].h[1] = *(const v8bf*)(p + hk + 16);
    }
    #pragma unroll
    for (int tm = 0; tm < 2; ++tm)
      #pragma unroll
      for (int tn = 0; tn < 4; ++tn)
        acc[tm][tn] = wmma_bf16(a[tm], b[tn], acc[tm][tn]);

    __syncthreads();
  }

  // Epilogue: element (r,lane) of tile = (row = r + 8*(lane>>4), col = lane&15)
  #pragma unroll
  for (int tm = 0; tm < 2; ++tm)
    #pragma unroll
    for (int tn = 0; tn < 4; ++tn) {
      const int colg = bN + nBase + tn * 16 + l16;
      #pragma unroll
      for (int r = 0; r < 8; ++r) {
        const int rowg = bM + mBase + tm * 16 + (lane >> 4) * 8 + r;
        const size_t off = (size_t)rowg * D + colg;
        float v = acc[tm][tn][r];
        if (epi == 0)      ((__bf16*)Out)[off] = (__bf16)v;
        else if (epi == 1) ((__bf16*)Out)[off] = (__bf16)(1.f / (1.f + __expf(-v)));
        else               ((float*)Out)[off]  = v + resid[off];
      }
    }
}

// ---------------------------------------------------------------------------
// Per-chunk attention: S = (Qc Kc^T) * mask ; O = S Vc ; Z = gate * O  (bf16)
// One workgroup per 32-row chunk (1024 WGs). D-reduction for S split over the
// 8 waves, merged with ds_add_f32 LDS atomics.
// ---------------------------------------------------------------------------
__global__ __launch_bounds__(256) void attn_chunk(const __bf16* __restrict__ Q,
                                                  const __bf16* __restrict__ Km,
                                                  const __bf16* __restrict__ V,
                                                  const __bf16* __restrict__ G,
                                                  const float* __restrict__ logDecay,
                                                  __bf16* __restrict__ Z) {
  __shared__ float  Sred[32][33];
  __shared__ __bf16 Sbf[32][LD];
  __shared__ __bf16 Vt[128][LD];   // transposed V slice: Vt[n][j]

  const int t    = threadIdx.x;
  const int lane = t & 31;
  const int w    = t >> 5;
  const int l16  = lane & 15;
  const int hk   = (lane >> 4) * 8;
  const int ks   = (lane >> 4) * 16;
  const int R    = blockIdx.x * 32;          // chunk row base

  for (int e = t; e < 32 * 33; e += 256) ((float*)Sred)[e] = 0.f;
  __syncthreads();

  // --- stage 1: partial S over this wave's D slice [w*128, w*128+128) ------
  v8f s[2][2] = {};
  for (int kk = 0; kk < 4; ++kk) {
    const int k0 = w * 128 + kk * 32;
    Frag a[2], b[2];
    #pragma unroll
    for (int tm = 0; tm < 2; ++tm) {
      const __bf16* p = Q + (size_t)(R + tm * 16 + l16) * D + k0;
      a[tm].h[0] = *(const v8bf*)(p + hk);
      a[tm].h[1] = *(const v8bf*)(p + hk + 16);
    }
    #pragma unroll
    for (int tn = 0; tn < 2; ++tn) {   // "columns" of S are K-rows
      const __bf16* p = Km + (size_t)(R + tn * 16 + l16) * D + k0;
      b[tn].h[0] = *(const v8bf*)(p + hk);
      b[tn].h[1] = *(const v8bf*)(p + hk + 16);
    }
    #pragma unroll
    for (int tm = 0; tm < 2; ++tm)
      #pragma unroll
      for (int tn = 0; tn < 2; ++tn)
        s[tm][tn] = wmma_bf16(a[tm], b[tn], s[tm][tn]);
  }
  #pragma unroll
  for (int tm = 0; tm < 2; ++tm)
    #pragma unroll
    for (int tn = 0; tn < 2; ++tn)
      #pragma unroll
      for (int r = 0; r < 8; ++r) {
        const int i = tm * 16 + (lane >> 4) * 8 + r;
        const int j = tn * 16 + l16;
        atomicAdd(&Sred[i][j], s[tm][tn][r]);     // ds_add_f32
      }
  __syncthreads();

  // --- decay mask + bf16 repack --------------------------------------------
  const float decay = 1.f / (1.f + __expf(-logDecay[0]));
  const float l2d   = __log2f(decay);
  for (int e = t; e < 1024; e += 256) {
    const int i = e >> 5, j = e & 31;
    const float m = (i >= j) ? exp2f(l2d * (float)(i - j)) : 0.f;
    Sbf[i][j] = (__bf16)(Sred[i][j] * m);
  }
  __syncthreads();

  Frag aS[2];
  #pragma unroll
  for (int tm = 0; tm < 2; ++tm) {
    const __bf16* p = &Sbf[tm * 16 + l16][0];
    aS[tm].h[0] = *(const v8bf*)(p + hk);
    aS[tm].h[1] = *(const v8bf*)(p + hk + 16);
  }

  // --- stage 3: O = S @ V in 8 column slices of 128; Z = gate*O ------------
  for (int ns = 0; ns < 8; ++ns) {
    const int nb = ns * 128;
    for (int e = t; e < 32 * 128; e += 256) {  // transpose V slice into LDS
      const int j = e >> 7, n = e & 127;
      Vt[n][j] = V[(size_t)(R + j) * D + nb + n];
    }
    __syncthreads();

    Frag bV;                                   // wave w owns 16-col subtile w
    const __bf16* p = &Vt[w * 16 + l16][0];
    bV.h[0] = *(const v8bf*)(p + hk);
    bV.h[1] = *(const v8bf*)(p + hk + 16);

    #pragma unroll
    for (int tm = 0; tm < 2; ++tm) {
      v8f o = {};
      o = wmma_bf16(aS[tm], bV, o);
      const int colg = nb + w * 16 + l16;
      #pragma unroll
      for (int r = 0; r < 8; ++r) {
        const int rowg = R + tm * 16 + (lane >> 4) * 8 + r;
        const size_t off = (size_t)rowg * D + colg;
        Z[off] = (__bf16)((float)G[off] * o[r]);
      }
    }
    __syncthreads();
  }
}

// ---------------------------------------------------------------------------
extern "C" void kernel_launch(void* const* d_in, const int* in_sizes, int n_in,
                              void* d_out, int out_size, void* d_ws, size_t ws_size,
                              hipStream_t stream) {
  const float* x   = (const float*)d_in[0];
  const float* Wsrc[5] = { (const float*)d_in[1], (const float*)d_in[2],
                           (const float*)d_in[3], (const float*)d_in[4],
                           (const float*)d_in[5] };
  const float* ldc = (const float*)d_in[6];
  float* out = (float*)d_out;

  char* ws = (char*)d_ws;
  size_t off = 0;
  __bf16* xb = (__bf16*)(ws + off); off += (size_t)M * D * 2;
  __bf16* Wb[5];
  for (int i = 0; i < 5; ++i) { Wb[i] = (__bf16*)(ws + off); off += (size_t)D * D * 2; }
  __bf16* Qb = (__bf16*)(ws + off); off += (size_t)M * D * 2;
  __bf16* Kb = (__bf16*)(ws + off); off += (size_t)M * D * 2;
  __bf16* Vb = (__bf16*)(ws + off); off += (size_t)M * D * 2;
  __bf16* Gb = (__bf16*)(ws + off); off += (size_t)M * D * 2;
  __bf16* Zb = (__bf16*)(ws + off); off += (size_t)M * D * 2;

  // 1) bf16 conversion
  cvt_bf16<<<(M * D) / 1024, 256, 0, stream>>>(x, xb, M * D);
  for (int i = 0; i < 5; ++i)
    cvt_bf16<<<(D * D) / 1024, 256, 0, stream>>>(Wsrc[i], Wb[i], D * D);

  // 2) projections
  dim3 g(D / 128, M / 128);
  gemm128<<<g, 256, 0, stream>>>(xb, Wb[0], nullptr, Qb, 0);
  gemm128<<<g, 256, 0, stream>>>(xb, Wb[1], nullptr, Kb, 0);
  gemm128<<<g, 256, 0, stream>>>(xb, Wb[2], nullptr, Vb, 0);
  gemm128<<<g, 256, 0, stream>>>(xb, Wb[3], nullptr, Gb, 1);  // sigmoid gate

  // 3) per-chunk masked attention + gating
  attn_chunk<<<M / 32, 256, 0, stream>>>(Qb, Kb, Vb, Gb, ldc, Zb);

  // 4) output projection + residual (f32 out)
  gemm128<<<g, 256, 0, stream>>>(Zb, Wb[4], x, out, 2);
}
